// SimpleEqDynamics_45707041964614
// MI455X (gfx1250) — matrix operand order repaired and checked by reading, not compiled, not run on hardware
//
#include <hip/hip_runtime.h>

typedef __attribute__((ext_vector_type(16))) _Float16 v16h;
typedef __attribute__((ext_vector_type(8)))  float    v8f;

#define N_PART 256
#define HIDDEN 128
#define N_RBF  64
#define BATCH  16
#define EPSQ   1e-6f

#if __has_builtin(__builtin_amdgcn_exp2f)
#define EXP2F(x) __builtin_amdgcn_exp2f(x)
#else
#define EXP2F(x) exp2f(x)
#endif

// ---------------------------------------------------------------------------
// Kernel 1: per-(batch, particle i) force accumulation.
// One workgroup = 128 threads = 4 waves. Each wave processes 4 tiles of 16
// j-particles. Per tile: two fused GEMMs (rbf@W1 and q@W1, q = d(rbf)/dd)
// via v_wmma_f32_16x16x32_f16, sharing B fragments of W1 staged in LDS.
// The N-tile loop is deliberately rolled (#pragma unroll 1) so only one
// (s,t) accumulator pair is live -> <256 VGPRs, no spills, no MSB traffic;
// cross-wave occupancy (4 waves/SIMD) fills the XDL pipe instead.
// ---------------------------------------------------------------------------
__global__ __launch_bounds__(128)
__attribute__((amdgpu_waves_per_eu(4)))
void eq_force_kernel(const float* __restrict__ xs,     // (B, 256*3)
                     const float* __restrict__ mus,    // (64,)
                     const float* __restrict__ loggam, // (64,)
                     const float* __restrict__ W1,     // (64,128) row-major
                     const float* __restrict__ b1,     // (128,)
                     const float* __restrict__ W2,     // (128,)
                     float* __restrict__ fws)          // (B,256,3) forces
{
    // 16 B-fragments (kt=2 K-tiles x nn=8 N-tiles), 32 lanes x 16 halves each
    __shared__ __align__(16) _Float16 ldsW[16 * 32 * 16]; // 16 KB
    // (mu, gamma*log2e) table indexed [hk][kt][hh] -> A-fragment K layout
    __shared__ float2 mgt[64];                            // 512 B
    // (b1[h], W2[h]) table, h = 0..127
    __shared__ float2 bwt[128];                           // 1 KB
    __shared__ float fred[4][3];

    const int i    = blockIdx.x;       // particle index
    const int b    = blockIdx.y;       // batch index
    const int tid  = threadIdx.x;
    const int lane = tid & 31;
    const int wave = tid >> 5;
    const int l16  = lane & 15;
    const int hk   = lane >> 4;        // which 16-lane half

    // ---- Stage W1 (f32 global) -> f16 B fragments in LDS.
    // B layout (32x16 KxN tile): lane gives column N=l16; halves give
    // K = kt*32 + hk*16 + hh  (K-striped across lane halves).
    for (int idx = tid; idx < 8192; idx += 128) {
        int hh   = idx & 15;
        int l    = (idx >> 4) & 31;
        int frag = idx >> 9;               // 0..15
        int kt   = frag >> 3;
        int nn   = frag & 7;
        int col  = nn * 16 + (l & 15);
        int K    = kt * 32 + (l >> 4) * 16 + hh;
        ldsW[idx] = (_Float16)W1[K * HIDDEN + col];
    }
    // ---- Stage per-lane-half (mu, gamma*log2e) table (A-fragment K mapping).
    if (tid < 64) {
        int hh  = tid & 15;
        int kt  = (tid >> 4) & 1;
        int hkI = tid >> 5;
        int K   = kt * 32 + hkI * 16 + hh;
        mgt[tid] = make_float2(mus[K],
                               __expf(loggam[K]) * 1.4426950408889634f);
    }
    // ---- Stage (b1, W2) table.
    bwt[tid] = make_float2(b1[tid], W2[tid]);

    const float xi0 = xs[b * (N_PART * 3) + i * 3 + 0];
    const float xi1 = xs[b * (N_PART * 3) + i * 3 + 1];
    const float xi2 = xs[b * (N_PART * 3) + i * 3 + 2];

    __syncthreads();   // LDS tables ready

    float fx = 0.f, fy = 0.f, fz = 0.f;

#pragma unroll 1
    for (int q = 0; q < 4; ++q) {
        const int jt = wave * 4 + q;       // tile of 16 j-values
        const int j  = jt * 16 + l16;      // j == i pad gives r=0 -> 0 force
        const float r0 = xi0 - xs[b * (N_PART * 3) + j * 3 + 0];
        const float r1 = xi1 - xs[b * (N_PART * 3) + j * 3 + 1];
        const float r2 = xi2 - xs[b * (N_PART * 3) + j * 3 + 2];
        const float d  = __builtin_amdgcn_sqrtf(r0 * r0 + r1 * r1 + r2 * r2 + EPSQ);

        // Build A fragments: rbf and its d-derivative q (both rows M=l16).
        // e = 2^(-(g2*dm)*dm),  q = -2ln2 * (g2*dm) * e   with g2 = gamma*log2e
        v16h Ar[2], Aq[2];
#pragma unroll
        for (int kt = 0; kt < 2; ++kt)
#pragma unroll
            for (int hh = 0; hh < 16; ++hh) {
                const float2 mg = mgt[(hk * 2 + kt) * 16 + hh];
                const float dm = d - mg.x;
                const float p1 = mg.y * dm;
                const float e  = EXP2F(-p1 * dm);
                const float qv = -1.3862943611198906f * (p1 * e);
                Ar[kt][hh] = (_Float16)e;
                Aq[kt][hh] = (_Float16)qv;
            }

        float gacc[8];
#pragma unroll
        for (int rr = 0; rr < 8; ++rr) gacc[rr] = 0.f;

#pragma unroll 1
        for (int nn = 0; nn < 8; ++nn) {
            const float2 bw = bwt[nn * 16 + l16];   // (b1[h], W2[h])
            v8f s, t;
#pragma unroll
            for (int rr = 0; rr < 8; ++rr) { s[rr] = bw.x; t[rr] = 0.f; }

            const v16h B0 = *(const v16h*)(ldsW + ((0 * 8 + nn) * 32 + lane) * 16);
            s = __builtin_amdgcn_wmma_f32_16x16x32_f16(
                    false, Ar[0], false, B0, (short)0, s, false, false);
            t = __builtin_amdgcn_wmma_f32_16x16x32_f16(
                    false, Aq[0], false, B0, (short)0, t, false, false);
            const v16h B1 = *(const v16h*)(ldsW + ((1 * 8 + nn) * 32 + lane) * 16);
            s = __builtin_amdgcn_wmma_f32_16x16x32_f16(
                    false, Ar[1], false, B1, (short)0, s, false, false);
            t = __builtin_amdgcn_wmma_f32_16x16x32_f16(
                    false, Aq[1], false, B1, (short)0, t, false, false);

            // g += W2 * sech^2(s) * t  elementwise; D layout: N=l16, M=rr+8*hk
#pragma unroll
            for (int rr = 0; rr < 8; ++rr) {
#if __has_builtin(__builtin_amdgcn_tanhf)
                float th    = __builtin_amdgcn_tanhf(s[rr]);
                float sech2 = fmaf(-th, th, 1.f);
#else
                float ss    = fminf(fmaxf(s[rr], -20.f), 20.f);
                float e2    = __expf(2.f * ss);
                float u     = __builtin_amdgcn_rcpf(e2 + 1.f);
                float sech2 = 4.f * e2 * u * u;
#endif
                gacc[rr] += bw.y * sech2 * t[rr];
            }
        }

        // Reduce over hidden columns (16-lane groups; xor stays in-group).
#pragma unroll
        for (int rr = 0; rr < 8; ++rr) {
            float gv = gacc[rr];
            gv += __shfl_xor(gv, 1, 32);
            gv += __shfl_xor(gv, 2, 32);
            gv += __shfl_xor(gv, 4, 32);
            gv += __shfl_xor(gv, 8, 32);
            gacc[rr] = gv;
        }
        // Route g of pair m=l16 to this lane. gacc[rr] in half hk == pair rr+8*hk.
        const int sel = l16 & 7;
        float v = gacc[0];
        v = (sel == 1) ? gacc[1] : v;
        v = (sel == 2) ? gacc[2] : v;
        v = (sel == 3) ? gacc[3] : v;
        v = (sel == 4) ? gacc[4] : v;
        v = (sel == 5) ? gacc[5] : v;
        v = (sel == 6) ? gacc[6] : v;
        v = (sel == 7) ? gacc[7] : v;
        const float g = __shfl(v, (((l16 >> 3) & 1) << 4) | (l16 & 7), 32);

        // Both lane-halves accumulate the same pair -> ordered-pair factor 2.
        const float sc = -g * __builtin_amdgcn_rcpf(d);
        fx += sc * r0;
        fy += sc * r1;
        fz += sc * r2;
    }

    // Wave-wide sum over all 32 lanes (pairs + the factor-2 duplication).
#pragma unroll
    for (int m = 1; m < 32; m <<= 1) {
        fx += __shfl_xor(fx, m, 32);
        fy += __shfl_xor(fy, m, 32);
        fz += __shfl_xor(fz, m, 32);
    }
    if (lane == 0) { fred[wave][0] = fx; fred[wave][1] = fy; fred[wave][2] = fz; }
    __syncthreads();
    if (tid == 0) {
        float ox = 0.f, oy = 0.f, oz = 0.f;
        for (int w = 0; w < 4; ++w) { ox += fred[w][0]; oy += fred[w][1]; oz += fred[w][2]; }
        float* o = fws + (b * N_PART + i) * 3;
        o[0] = ox; o[1] = oy; o[2] = oz;
    }
}

// ---------------------------------------------------------------------------
// Kernel 2: subtract per-batch mean over particles (zero CoM drift).
// ---------------------------------------------------------------------------
__global__ __launch_bounds__(256)
void meansub_kernel(const float* __restrict__ fws, float* __restrict__ out)
{
    __shared__ float sx[256], sy[256], sz[256];
    const int b = blockIdx.x;
    const int n = threadIdx.x;
    const float* p = fws + (b * N_PART + n) * 3;
    const float fx = p[0], fy = p[1], fz = p[2];
    sx[n] = fx; sy[n] = fy; sz[n] = fz;
    __syncthreads();
#pragma unroll
    for (int off = 128; off > 0; off >>= 1) {
        if (n < off) { sx[n] += sx[n + off]; sy[n] += sy[n + off]; sz[n] += sz[n + off]; }
        __syncthreads();
    }
    const float inv = 1.0f / 256.0f;
    const float mx = sx[0] * inv, my = sy[0] * inv, mz = sz[0] * inv;
    float* o = out + (b * N_PART + n) * 3;
    o[0] = fx - mx; o[1] = fy - my; o[2] = fz - mz;
}

extern "C" void kernel_launch(void* const* d_in, const int* in_sizes, int n_in,
                              void* d_out, int out_size, void* d_ws, size_t ws_size,
                              hipStream_t stream) {
    // setup_inputs order: t, xs, mus, log_gammas, W1, b1, W2, b2 (all f32)
    const float* xs  = (const float*)d_in[1];
    const float* mus = (const float*)d_in[2];
    const float* lg  = (const float*)d_in[3];
    const float* W1  = (const float*)d_in[4];
    const float* b1  = (const float*)d_in[5];
    const float* W2  = (const float*)d_in[6];
    float* fws = (float*)d_ws;                 // (16,256,3) = 48 KB scratch
    float* out = (float*)d_out;

    eq_force_kernel<<<dim3(N_PART, BATCH), 128, 0, stream>>>(
        xs, mus, lg, W1, b1, W2, fws);
    meansub_kernel<<<BATCH, N_PART, 0, stream>>>(fws, out);
}
